// DeltaNet_31877247271553
// MI455X (gfx1250) — matrix operand order repaired and checked by reading, not compiled, not run on hardware
//
#include <hip/hip_runtime.h>

typedef _Float16 half_t;
typedef __attribute__((ext_vector_type(16))) _Float16 v16h;
typedef __attribute__((ext_vector_type(2)))  _Float16 v2h;
typedef __attribute__((ext_vector_type(8)))  float    v8f;
typedef __attribute__((ext_vector_type(4)))  unsigned int v4u;
typedef __attribute__((ext_vector_type(8)))  int      v8i;
typedef __attribute__((ext_vector_type(4)))  int      v4i;

#define BB 2
#define HH 8
#define LL 4096
#define DD 128
#define CC 32
#define NCHUNK (LL / CC)
#define NCB 4                // column-blocks over d_v
#define VC  (DD / NCB)       // 32 v-columns per block
#define LDH 136              // f16 LDS row pitch (halves)
#define LDF 33               // f32 LDS row pitch (floats)

__device__ __forceinline__ v8f wmma16(v16h a, v16h b, v8f c) {
  return __builtin_amdgcn_wmma_f32_16x16x32_f16(false, a, false, b, (short)0, c,
                                                false, false);
}

// ---- fragment gathers (wave32 WMMA, contiguous k = 16*(lane/16) + e in both A and B) ----------

// A (16x32) from f16 row-major [rows][ld]
__device__ __forceinline__ v16h ldA_h(const half_t* base, int ld, int m0, int k0) {
  const int l = threadIdx.x & 31;
  const int m = m0 + (l & 15);
  const int g = (l >> 4) << 3;
  v16h a;
#pragma unroll
  for (int j = 0; j < 8; ++j) {
    const int e = 2 * j;
    const int kl = 2 * g + e;   // even kl, pair-contiguous
    v2h pr = *(const v2h*)&base[m * ld + k0 + kl];
    a[e] = pr[0];
    a[e + 1] = pr[1];
  }
  return a;
}

// A (16x32) transposed source: A[m][k] = base[k][m]   (f16)
__device__ __forceinline__ v16h ldA_hT(const half_t* base, int ld, int m0, int k0) {
  const int l = threadIdx.x & 31;
  const int m = m0 + (l & 15);
  const int g = (l >> 4) << 3;
  v16h a;
#pragma unroll
  for (int e = 0; e < 16; ++e) {
    const int kl = 2 * g + e;
    a[e] = base[(k0 + kl) * ld + m];
  }
  return a;
}

// A (16x32) from f32 row-major, zero-masked at kl >= kmax
__device__ __forceinline__ v16h ldA_f(const float* base, int ld, int m0, int k0,
                                      int kmax) {
  const int l = threadIdx.x & 31;
  const int m = m0 + (l & 15);
  const int g = (l >> 4) << 3;
  v16h a;
#pragma unroll
  for (int e = 0; e < 16; ++e) {
    const int kl = 2 * g + e;
    a[e] = (kl < kmax) ? (half_t)base[m * ld + k0 + kl] : (half_t)0.0f;
  }
  return a;
}

// B (32x16) from f32 row-major [k][n], zero-masked at kl >= kmax
__device__ __forceinline__ v16h ldB_f(const float* base, int ld, int k0, int n0,
                                      int kmax) {
  const int l = threadIdx.x & 31;
  const int n = n0 + (l & 15);
  const int kb = (l >> 4) << 4;
  v16h b;
#pragma unroll
  for (int e = 0; e < 16; ++e) {
    const int kl = kb + e;
    b[e] = (kl < kmax) ? (half_t)base[(k0 + kl) * ld + n] : (half_t)0.0f;
  }
  return b;
}

// B (32x16) from f16 row-major [k][n]
__device__ __forceinline__ v16h ldB_h(const half_t* base, int ld, int k0, int n0) {
  const int l = threadIdx.x & 31;
  const int n = n0 + (l & 15);
  const int kb = (l >> 4) << 4;
  v16h b;
#pragma unroll
  for (int e = 0; e < 16; ++e) b[e] = base[(k0 + kb + e) * ld + n];
  return b;
}

// B (32x16) transposed source: B[k][n] = base[n][k]   (f16)
__device__ __forceinline__ v16h ldB_hT(const half_t* base, int ld, int k0, int n0) {
  const int l = threadIdx.x & 31;
  const int n = n0 + (l & 15);
  const int kb = (l >> 4) << 4;
  v16h b;
#pragma unroll
  for (int e = 0; e < 16; ++e) b[e] = base[n * ld + k0 + kb + e];
  return b;
}

// C/D (16x16 f32) <-> LDS / global
__device__ __forceinline__ v8f ldC_f(const float* base, int ld, int m0, int n0) {
  const int l = threadIdx.x & 31;
  const int n = n0 + (l & 15);
  const int mb = m0 + ((l >> 4) << 3);
  v8f c;
#pragma unroll
  for (int r = 0; r < 8; ++r) c[r] = base[(mb + r) * ld + n];
  return c;
}
__device__ __forceinline__ void stC_f(float* base, int ld, int m0, int n0, v8f c) {
  const int l = threadIdx.x & 31;
  const int n = n0 + (l & 15);
  const int mb = m0 + ((l >> 4) << 3);
#pragma unroll
  for (int r = 0; r < 8; ++r) base[(mb + r) * ld + n] = c[r];
}
__device__ __forceinline__ void stC_h(half_t* base, int ld, int m0, int n0, v8f c) {
  const int l = threadIdx.x & 31;
  const int n = n0 + (l & 15);
  const int mb = m0 + ((l >> 4) << 3);
#pragma unroll
  for (int r = 0; r < 8; ++r) base[(mb + r) * ld + n] = (half_t)c[r];
}
__device__ __forceinline__ void stC_causal(float* base, int ld, int m0, int n0,
                                           v8f c) {
  const int l = threadIdx.x & 31;
  const int n = n0 + (l & 15);
  const int mb = m0 + ((l >> 4) << 3);
#pragma unroll
  for (int r = 0; r < 8; ++r)
    base[(mb + r) * ld + n] = (n <= (mb + r)) ? c[r] : 0.0f;
}
__device__ __forceinline__ void stC_g(float* base, int ld, int m0, int n0, v8f c) {
  const int l = threadIdx.x & 31;
  const int n = n0 + (l & 15);
  const int mb = m0 + ((l >> 4) << 3);
#pragma unroll
  for (int r = 0; r < 8; ++r) base[(size_t)(mb + r) * ld + n] = c[r];
}

// Forward substitution: invert unit-lower 16x16 block of T (strict entries in
// sT, implicit unit diagonal) at (r0,r0) into X. Lanes 0..15 own columns.
__device__ __forceinline__ void fwd_subst16(const float* T, float* X, int r0) {
  const int l = threadIdx.x & 31;
  if (l < 16) {
    const int c = l;
    for (int r = 0; r < 16; ++r) {
      float acc = (r == c) ? 1.0f : 0.0f;
      for (int j = 0; j < r; ++j)
        acc -= T[(r0 + r) * LDF + (r0 + j)] * X[(r0 + j) * LDF + (r0 + c)];
      X[(r0 + r) * LDF + (r0 + c)] = acc;
    }
  }
}

__global__ __launch_bounds__(256) void deltanet_scan_kernel(
    const float* __restrict__ q, const float* __restrict__ k,
    const float* __restrict__ v, const float* __restrict__ beta,
    float* __restrict__ out) {
  __shared__ half_t sQ[CC * LDH];    // normalized q (f16)
  __shared__ half_t sK[CC * LDH];    // normalized k (f16)
  __shared__ half_t sKB[CC * LDH];   // k*beta, then w = T^-1 @ (k*beta)
  __shared__ float  sV[CC * LDF];    // v*beta slice -> u -> u_adj (f32)
  __shared__ float  sT[CC * LDF];    // TDM stage for v-tile, then A = kb @ k^T
  __shared__ float  sInv[CC * LDF];  // T^-1
  __shared__ float  sAttn[CC * LDF]; // causal(q @ k^T)
  __shared__ float  sS[DD * LDF];    // state slice, 128 x 32 (f32)
  __shared__ float  sTmp[16 * 17];

  const int bh = blockIdx.x / NCB;
  const int vcol0 = (blockIdx.x % NCB) * VC;
  const int tid = threadIdx.x;
  const int wave = tid >> 5;
  const int lane = tid & 31;

  for (int i = tid; i < DD * LDF; i += 256) sS[i] = 0.0f;
  __syncthreads();

  const size_t baseQK = (size_t)bh * LL * DD;
  float* outO = out + baseQK;
  float* outS = out + (size_t)BB * HH * LL * DD + (size_t)bh * DD * DD;

  for (int ch = 0; ch < NCHUNK; ++ch) {
    // ---------- phase 0a: TDM the 32x32 v-tile (strided in a 128-wide tensor)
    // into the (currently dead) sT buffer; overlaps the q/k loads + norms.
    if (wave == 0) {
      const unsigned lds_off = (unsigned)(size_t)(void*)sT;
      const unsigned long long ga = (unsigned long long)(size_t)(const void*)(
          v + baseQK + (size_t)(ch * CC) * DD + vcol0);
      v4u g0;
      g0[0] = 1u;                                   // count=1, user descriptor
      g0[1] = lds_off;                              // lds_addr (bytes)
      g0[2] = (unsigned)(ga & 0xFFFFFFFFu);         // global_addr[31:0]
      g0[3] = ((unsigned)(ga >> 32) & 0x01FFFFFFu)  // global_addr[56:32]
              | (2u << 30);                         // type = 2 ("image")
      v8i g1;
      g1[0] = (int)(2u << 16);        // wg_mask=0, data_size=2 (4 bytes)
      g1[1] = (int)((unsigned)DD << 16);  // tensor_dim0[15:0] = 128
      g1[2] = (int)((unsigned)CC << 16);  // tensor_dim0[31:16]=0, tensor_dim1=32
      g1[3] = (int)((unsigned)VC << 16);  // tensor_dim1 hi=0, tile_dim0 = 32
      g1[4] = (int)CC;                    // tile_dim1 = 32, tile_dim2 = 0
      g1[5] = (int)DD;                    // tensor_dim0_stride = 128 (lo32)
      g1[6] = 0;                          // stride0 hi, stride1 lo
      g1[7] = 0;                          // stride1 hi
      v4i g2 = {0, 0, 0, 0};
      v4i g3 = {0, 0, 0, 0};
      v8i g4 = {0, 0, 0, 0, 0, 0, 0, 0};
      __builtin_amdgcn_tensor_load_to_lds(g0, g1, g2, g3, g4, 0);
    }

    // ---------- phase 0b: load q/k chunk, l2norm, scale by beta ----------
    const int row = tid >> 3;       // 32 rows, 8 lanes per row
    const int sub = tid & 7;
    float bet;
    {
      const int c0 = sub * 16;
      const size_t roff = baseQK + (size_t)(ch * CC + row) * DD;
      float qv[16], kv[16];
      float sq = 0.0f, sk = 0.0f;
#pragma unroll
      for (int i = 0; i < 16; ++i) {
        qv[i] = q[roff + c0 + i]; sq += qv[i] * qv[i];
        kv[i] = k[roff + c0 + i]; sk += kv[i] * kv[i];
      }
#pragma unroll
      for (int m = 1; m < 8; m <<= 1) {
        sq += __shfl_xor(sq, m, 32);
        sk += __shfl_xor(sk, m, 32);
      }
      const float rq = rsqrtf(sq), rk = rsqrtf(sk);
      bet = beta[(size_t)bh * LL + ch * CC + row];
#pragma unroll
      for (int i = 0; i < 16; ++i) {
        sQ[row * LDH + c0 + i] = (half_t)(qv[i] * rq);
        const float kn = kv[i] * rk;
        sK[row * LDH + c0 + i] = (half_t)kn;
        sKB[row * LDH + c0 + i] = (half_t)(kn * bet);
      }
      if (ch + 1 < NCHUNK) {   // pull next chunk toward L2/L0
        __builtin_prefetch(&q[roff + (size_t)CC * DD + c0], 0, 1);
        __builtin_prefetch(&k[roff + (size_t)CC * DD + c0], 0, 1);
      }
    }
    // ---------- phase 0c: wait for TDM, then beta-scale v-tile into sV
    if (wave == 0) __builtin_amdgcn_s_wait_tensorcnt(0);
    __syncthreads();
#pragma unroll
    for (int i = 0; i < 4; ++i)
      sV[row * LDF + sub * 4 + i] = sT[row * VC + sub * 4 + i] * bet;
    __syncthreads();

    // ---------- phase 1: A = KB@K^T (w0-3) || attn = causal(Q@K^T) (w4-7)
    if (wave < 4) {
      const int mt = (wave >> 1) * 16, nt = (wave & 1) * 16;
      v8f acc = {};
#pragma unroll
      for (int kk = 0; kk < 4; ++kk)
        acc = wmma16(ldA_h(sKB, LDH, mt, kk * 32),
                     ldB_hT(sK, LDH, kk * 32, nt), acc);
      stC_f(sT, LDF, mt, nt, acc);
    } else {
      const int w2 = wave - 4;
      const int mt = (w2 >> 1) * 16, nt = (w2 & 1) * 16;
      v8f acc = {};
#pragma unroll
      for (int kk = 0; kk < 4; ++kk)
        acc = wmma16(ldA_h(sQ, LDH, mt, kk * 32),
                     ldB_hT(sK, LDH, kk * 32, nt), acc);
      stC_causal(sAttn, LDF, mt, nt, acc);
    }
    __syncthreads();

    // ---------- phase 2a: invert diagonal 16x16 blocks (w0,w1), zero X12 (w2)
    if (wave == 0) {
      fwd_subst16(sT, sInv, 0);
    } else if (wave == 1) {
      fwd_subst16(sT, sInv, 16);
    } else if (wave == 2) {
      for (int i = lane; i < 256; i += 32)
        sInv[(i >> 4) * LDF + 16 + (i & 15)] = 0.0f;
    }
    __syncthreads();
    // ---------- phase 2b: X21 = -X22 @ A21 @ X11 (wave 0, WMMA, K masked to 16)
    if (wave == 0) {
      v8f z = {};
      v8f t1 = wmma16(ldA_f(sT, LDF, 16, 0, 16), ldB_f(sInv, LDF, 0, 0, 16), z);
      stC_f(sTmp, 17, 0, 0, t1);
      v8f z2 = {};
      v8f x21 = wmma16(ldA_f(sInv, LDF, 16, 16, 16),
                       ldB_f(sTmp, 17, 0, 0, 16), z2);
#pragma unroll
      for (int r = 0; r < 8; ++r) x21[r] = -x21[r];
      stC_f(sInv, LDF, 16, 0, x21);
    }
    __syncthreads();

    // ---------- phase 3: w = T^-1 @ KB (all waves) ; u = T^-1 @ V (w0-3)
    v8f wacc0 = {}, wacc1 = {}, uacc = {};
    {
      const int t0 = wave * 2, t1i = wave * 2 + 1;
      wacc0 = wmma16(ldA_f(sInv, LDF, (t0 >> 3) * 16, 0, 32),
                     ldB_h(sKB, LDH, 0, (t0 & 7) * 16), wacc0);
      wacc1 = wmma16(ldA_f(sInv, LDF, (t1i >> 3) * 16, 0, 32),
                     ldB_h(sKB, LDH, 0, (t1i & 7) * 16), wacc1);
      if (wave < 4)
        uacc = wmma16(ldA_f(sInv, LDF, (wave >> 1) * 16, 0, 32),
                      ldB_f(sV, LDF, 0, (wave & 1) * 16, 32), uacc);
    }
    __syncthreads();   // all reads of kb/v done before overwrite
    {
      const int t0 = wave * 2, t1i = wave * 2 + 1;
      stC_h(sKB, LDH, (t0 >> 3) * 16, (t0 & 7) * 16, wacc0);
      stC_h(sKB, LDH, (t1i >> 3) * 16, (t1i & 7) * 16, wacc1);
      if (wave < 4) stC_f(sV, LDF, (wave >> 1) * 16, (wave & 1) * 16, uacc);
    }
    __syncthreads();

    // ---------- phase 5: u_adj = u - w@S (w0-3) || accO = Q@S (w4-7)
    v8f accO = {};
    if (wave < 4) {
      const int mt = (wave >> 1) * 16, nt = (wave & 1) * 16;
      v8f acc = {};
#pragma unroll
      for (int kk = 0; kk < 4; ++kk)
        acc = wmma16(ldA_h(sKB, LDH, mt, kk * 32),
                     ldB_f(sS, LDF, kk * 32, nt, 32), acc);
      v8f cu = ldC_f(sV, LDF, mt, nt);
#pragma unroll
      for (int r = 0; r < 8; ++r) cu[r] -= acc[r];
      stC_f(sV, LDF, mt, nt, cu);   // in-place, tile-disjoint
    } else {
      const int w2 = wave - 4;
      const int mt = (w2 >> 1) * 16, nt = (w2 & 1) * 16;
#pragma unroll
      for (int kk = 0; kk < 4; ++kk)
        accO = wmma16(ldA_h(sQ, LDH, mt, kk * 32),
                      ldB_f(sS, LDF, kk * 32, nt, 32), accO);
    }
    __syncthreads();

    // ---------- phase 6: O += attn@u_adj -> global (w4-7) || S += K^T@u_adj (w0-3)
    if (wave >= 4) {
      const int w2 = wave - 4;
      const int mt = (w2 >> 1) * 16, nt = (w2 & 1) * 16;
      accO = wmma16(ldA_f(sAttn, LDF, mt, 0, 32),
                    ldB_f(sV, LDF, 0, nt, 32), accO);
      stC_g(outO + (size_t)(ch * CC) * DD + vcol0, DD, mt, nt, accO);
    } else {
#pragma unroll
      for (int i = 0; i < 4; ++i) {
        const int tt = wave * 4 + i;
        const int mt = (tt >> 1) * 16, nt = (tt & 1) * 16;
        v8f cs = ldC_f(sS, LDF, mt, nt);
        cs = wmma16(ldA_hT(sK, LDH, mt, 0), ldB_f(sV, LDF, 0, nt, 32), cs);
        stC_f(sS, LDF, mt, nt, cs);
      }
    }
    __syncthreads();
  }

  // ---------- final state write-out ----------
  for (int i = tid; i < DD * VC; i += 256) {
    const int m = i >> 5, n = i & 31;
    outS[(size_t)m * DD + vcol0 + n] = sS[m * LDF + n];
  }
}

extern "C" void kernel_launch(void* const* d_in, const int* in_sizes, int n_in,
                              void* d_out, int out_size, void* d_ws,
                              size_t ws_size, hipStream_t stream) {
  (void)in_sizes; (void)n_in; (void)out_size; (void)d_ws; (void)ws_size;
  const float* q = (const float*)d_in[0];
  const float* k = (const float*)d_in[1];
  const float* v = (const float*)d_in[2];
  const float* beta = (const float*)d_in[3];
  float* out = (float*)d_out;
  dim3 grid(BB * HH * NCB);
  dim3 block(256);
  deltanet_scan_kernel<<<grid, block, 0, stream>>>(q, k, v, beta, out);
}
